// GATNet_35966056137388
// MI455X (gfx1250) — matrix operand order repaired and checked by reading, not compiled, not run on hardware
//
#include <hip/hip_runtime.h>
#include <hip/hip_bf16.h>

typedef __attribute__((ext_vector_type(16))) __bf16 v16bf;
typedef __attribute__((ext_vector_type(8)))  __bf16 v8bf;
typedef __attribute__((ext_vector_type(4)))  __bf16 v4bf;
typedef __attribute__((ext_vector_type(8)))  float  v8f;

#define N_NODESC 50000
#define N_EDGESC 800000
#define E_TOTC   (N_EDGESC + N_NODESC)
#define N_GRAPHSC 128
#define F_INC 36
#define HEADS1C 10
#define D1C 360      // HEADS1*F_IN
#define D1PC 384     // padded feature width layer 1
#define OUTC 128

#define AST 40       // LDS row stride (bf16) for A tile: 80B, 16B-aligned, conflict-free
#define BST 40       // LDS row stride (bf16) for B tile

// ---------------- float atomic max via sign-split integer atomics ----------
__device__ inline void atomicMaxF(float* addr, float val) {
    if (val >= 0.0f) atomicMax((int*)addr, __float_as_int(val));
    else             atomicMin((unsigned int*)addr, __float_as_uint(val));
}

// ---------------- generic utility kernels ----------------------------------
__global__ void fill_f32(float* __restrict__ p, float v, long n) {
    long idx = (long)blockIdx.x * blockDim.x + threadIdx.x;
    if (idx < n) p[idx] = v;
}

// f32 [K,N] row-major -> bf16 TRANSPOSED [Np,Kp] (n-major), zero-padded
__global__ void pad_convert_T(const float* __restrict__ src, __bf16* __restrict__ dst,
                              int K, int N, int Kp, int Np) {
    int idx = blockIdx.x * blockDim.x + threadIdx.x;
    if (idx >= Np * Kp) return;
    int n = idx / Kp, k = idx % Kp;
    float v = (k < K && n < N) ? src[k * N + n] : 0.0f;
    dst[idx] = (__bf16)v;
}

// ---------------- bf16 WMMA GEMM: C[M x Npad] = A[M x K] * BT[Npad x Kpad]^T
// Block tile 64(M) x 128(N), 8 waves: wave(wr,wc) -> rows 16*wr, cols 64*wc.
// Register-staged pipeline: iteration i commits regs->LDS, issues global loads
// for tile i+1, then runs 4 WMMAs against LDS while loads are in flight.
__global__ __launch_bounds__(256)
void gemm_bf16_wmma(const float* __restrict__ A, int lda,
                    const __bf16* __restrict__ BT, int Kpad,
                    float* __restrict__ C, int ldc,
                    int M, int K) {
    __shared__ __bf16 As[64 * AST];     // 64 rows (m) x 32 (k)
    __shared__ __bf16 Bs[128 * BST];    // 128 rows (n) x 32 (k)
    const int lane = threadIdx.x & 31;
    const int wave = threadIdx.x >> 5;
    const int wr = wave >> 1;
    const int wc = wave & 1;
    const int hi = lane >> 4;          // 16-lane half
    const int l16 = lane & 15;
    const int m0 = blockIdx.y * 64;
    const int n0 = blockIdx.x * 128;

    // per-thread staging registers (2 chunks each of A and B)
    float4 ra[2];
    v8bf   rb[2];

    // A chunk j: element i = tid + j*256 of 512; row = i>>3, k4 = (i&7)*4
    // B chunk j: row = i>>2, k8 = (i&3)*8
    auto prefetch = [&](int k0) {
#pragma unroll
        for (int j = 0; j < 2; ++j) {
            int i = threadIdx.x + j * 256;
            int rA = i >> 3, k4 = (i & 7) * 4;
            int row = m0 + rA, k = k0 + k4;
            float4 f = make_float4(0.f, 0.f, 0.f, 0.f);
            if (row < M && k < K) f = *(const float4*)(A + (long)row * lda + k);
            ra[j] = f;
            int rB = i >> 2, k8 = (i & 3) * 8;
            rb[j] = *(const v8bf*)(BT + (long)(n0 + rB) * Kpad + k0 + k8);
        }
    };
    auto commit = [&]() {
#pragma unroll
        for (int j = 0; j < 2; ++j) {
            int i = threadIdx.x + j * 256;
            int rA = i >> 3, k4 = (i & 7) * 4;
            v4bf pv = { (__bf16)ra[j].x, (__bf16)ra[j].y,
                        (__bf16)ra[j].z, (__bf16)ra[j].w };
            *(v4bf*)&As[rA * AST + k4] = pv;
            int rB = i >> 2, k8 = (i & 3) * 8;
            *(v8bf*)&Bs[rB * BST + k8] = rb[j];
        }
    };
    auto ldB = [&](int nrow) -> v16bf {
        const __bf16* brow = &Bs[nrow * BST + hi * 16];
        v8bf b0 = *(const v8bf*)brow;
        v8bf b1 = *(const v8bf*)(brow + 8);
        return __builtin_shufflevector(b0, b1, 0, 1, 2, 3, 4, 5, 6, 7,
                                       8, 9, 10, 11, 12, 13, 14, 15);
    };

    v8f acc0 = {}, acc1 = {}, acc2 = {}, acc3 = {};

    prefetch(0);
    for (int k0 = 0; k0 < Kpad; k0 += 32) {
        commit();                       // waits on outstanding global loads
        __syncthreads();
        if (k0 + 32 < Kpad) prefetch(k0 + 32);   // issue next tile's loads

        // A fragment: two contiguous b128 LDS loads, k = hi*16 + 0..15 (matches B)
        const __bf16* arow = &As[(wr * 16 + l16) * AST];
        v8bf a0 = *(const v8bf*)(arow + hi * 16);
        v8bf a1 = *(const v8bf*)(arow + hi * 16 + 8);
        v16bf fa = __builtin_shufflevector(a0, a1, 0, 1, 2, 3, 4, 5, 6, 7,
                                           8, 9, 10, 11, 12, 13, 14, 15);
        // B fragments batched: k = hi*16 + 0..15 contiguous in transposed tile
        v16bf fb0 = ldB(wc * 64 + l16);
        v16bf fb1 = ldB(wc * 64 + 16 + l16);
        v16bf fb2 = ldB(wc * 64 + 32 + l16);
        v16bf fb3 = ldB(wc * 64 + 48 + l16);
        acc0 = __builtin_amdgcn_wmma_f32_16x16x32_bf16(false, fa, false, fb0,
                                                       (short)0, acc0, false, false);
        acc1 = __builtin_amdgcn_wmma_f32_16x16x32_bf16(false, fa, false, fb1,
                                                       (short)0, acc1, false, false);
        acc2 = __builtin_amdgcn_wmma_f32_16x16x32_bf16(false, fa, false, fb2,
                                                       (short)0, acc2, false, false);
        acc3 = __builtin_amdgcn_wmma_f32_16x16x32_bf16(false, fa, false, fb3,
                                                       (short)0, acc3, false, false);
        __syncthreads();
    }

    // C/D layout: lane -> n = l16 ; vgpr r -> m = r + 8*hi
    const int rbase = m0 + wr * 16 + hi * 8;
    float* cp = C + (long)rbase * ldc + n0 + wc * 64 + l16;
#pragma unroll
    for (int r = 0; r < 8; ++r) {
        if (rbase + r < M) {
            cp[0]  = acc0[r];
            cp[16] = acc1[r];
            cp[32] = acc2[r];
            cp[48] = acc3[r];
        }
        cp += ldc;
    }
}

// ---------------- attention logits ------------------------------------------
__global__ void attn_logits1(const float* __restrict__ h, const float* __restrict__ asrc,
                             const float* __restrict__ adst,
                             float* __restrict__ als, float* __restrict__ ald) {
    int idx = blockIdx.x * blockDim.x + threadIdx.x;
    if (idx >= N_NODESC * HEADS1C) return;
    int n = idx / HEADS1C, hh = idx % HEADS1C;
    const float* row = h + (long)n * D1PC + hh * F_INC;
    const float* as = asrc + hh * F_INC;
    const float* ad = adst + hh * F_INC;
    float ss = 0.0f, dd = 0.0f;
#pragma unroll
    for (int c = 0; c < F_INC; ++c) { float v = row[c]; ss += v * as[c]; dd += v * ad[c]; }
    als[idx] = ss; ald[idx] = dd;
}

__global__ void attn_logits2(const float* __restrict__ h, const float* __restrict__ asrc,
                             const float* __restrict__ adst,
                             float* __restrict__ als, float* __restrict__ ald) {
    int n = blockIdx.x * blockDim.x + threadIdx.x;
    if (n >= N_NODESC) return;
    const float* row = h + (long)n * OUTC;
    float ss = 0.0f, dd = 0.0f;
    for (int c = 0; c < OUTC; ++c) { float v = row[c]; ss += v * asrc[c]; dd += v * adst[c]; }
    als[n] = ss; ald[n] = dd;
}

// ---------------- edge phase, layer 1 ---------------------------------------
__global__ void edge_max1(const int* __restrict__ ei, const float* __restrict__ als,
                          const float* __restrict__ ald, float* __restrict__ m) {
    long idx = (long)blockIdx.x * blockDim.x + threadIdx.x;
    if (idx >= (long)E_TOTC * HEADS1C) return;
    int e = (int)(idx / HEADS1C), hh = (int)(idx % HEADS1C);
    int s, d;
    if (e < N_EDGESC) { s = ei[e]; d = ei[N_EDGESC + e]; } else { s = d = e - N_EDGESC; }
    float v = als[s * HEADS1C + hh] + ald[d * HEADS1C + hh];
    v = v > 0.0f ? v : 0.2f * v;            // leaky_relu 0.2
    atomicMaxF(&m[d * HEADS1C + hh], v);
}

// wave-per-edge: unnormalized softmax weight p per head, scatter p*h[src] + p
__global__ void edge_pass1(const int* __restrict__ ei, const float* __restrict__ als,
                           const float* __restrict__ ald, const float* __restrict__ m,
                           const float* __restrict__ h, float* __restrict__ ssum,
                           float* __restrict__ acc) {
    int gw = blockIdx.x * (blockDim.x >> 5) + (threadIdx.x >> 5);
    int lane = threadIdx.x & 31;
    if (gw >= E_TOTC) return;
    int s, d;
    if (gw < N_EDGESC) { s = ei[gw]; d = ei[N_EDGESC + gw]; } else { s = d = gw - N_EDGESC; }
    float p = 0.0f;
    if (lane < HEADS1C) {
        float v = als[s * HEADS1C + lane] + ald[d * HEADS1C + lane];
        v = v > 0.0f ? v : 0.2f * v;
        p = __expf(v - m[d * HEADS1C + lane]);
        atomicAdd(&ssum[d * HEADS1C + lane], p);
    }
    const float* hs = h + (long)s * D1PC;
    float* ac = acc + (long)d * D1PC;
    __builtin_prefetch(hs, 0, 0);
    for (int c = lane; c < D1C; c += 32) {
        float pc = __shfl(p, c / F_INC, 32);
        atomicAdd(&ac[c], pc * hs[c]);
    }
}

__global__ void finalize1(float* __restrict__ acc, const float* __restrict__ ssum,
                          const float* __restrict__ bias) {
    long idx = (long)blockIdx.x * blockDim.x + threadIdx.x;
    if (idx >= (long)N_NODESC * D1C) return;
    int n = (int)(idx / D1C), c = (int)(idx % D1C);
    float v = acc[(long)n * D1PC + c] / (ssum[n * HEADS1C + c / F_INC] + 1e-16f) + bias[c];
    v = v > 0.0f ? v : (__expf(v) - 1.0f);  // ELU
    acc[(long)n * D1PC + c] = v;
}

// ---------------- edge phase, layer 2 (1 head, 128 ch) ----------------------
__global__ void edge_max2(const int* __restrict__ ei, const float* __restrict__ als,
                          const float* __restrict__ ald, float* __restrict__ m) {
    int e = blockIdx.x * blockDim.x + threadIdx.x;
    if (e >= E_TOTC) return;
    int s, d;
    if (e < N_EDGESC) { s = ei[e]; d = ei[N_EDGESC + e]; } else { s = d = e - N_EDGESC; }
    float v = als[s] + ald[d];
    v = v > 0.0f ? v : 0.2f * v;
    atomicMaxF(&m[d], v);
}

__global__ void edge_pass2(const int* __restrict__ ei, const float* __restrict__ als,
                           const float* __restrict__ ald, const float* __restrict__ m,
                           const float* __restrict__ h, float* __restrict__ ssum,
                           float* __restrict__ acc) {
    int gw = blockIdx.x * (blockDim.x >> 5) + (threadIdx.x >> 5);
    int lane = threadIdx.x & 31;
    if (gw >= E_TOTC) return;
    int s, d;
    if (gw < N_EDGESC) { s = ei[gw]; d = ei[N_EDGESC + gw]; } else { s = d = gw - N_EDGESC; }
    float p = 0.0f;
    if (lane == 0) {
        float v = als[s] + ald[d];
        v = v > 0.0f ? v : 0.2f * v;
        p = __expf(v - m[d]);
        atomicAdd(&ssum[d], p);
    }
    p = __shfl(p, 0, 32);
    const float* hs = h + (long)s * OUTC;
    float* ac = acc + (long)d * OUTC;
#pragma unroll
    for (int c = lane; c < OUTC; c += 32)
        atomicAdd(&ac[c], p * hs[c]);
}

__global__ void finalize2(float* __restrict__ acc, const float* __restrict__ ssum,
                          const float* __restrict__ bias) {
    long idx = (long)blockIdx.x * blockDim.x + threadIdx.x;
    if (idx >= (long)N_NODESC * OUTC) return;
    int n = (int)(idx / OUTC), c = (int)(idx % OUTC);
    float v = acc[idx] / (ssum[n] + 1e-16f) + bias[c];
    acc[idx] = v > 0.0f ? v : 0.0f;          // ReLU
}

// ---------------- pool + FC -------------------------------------------------
__global__ void pool_max(const float* __restrict__ h, const int* __restrict__ batch,
                         float* __restrict__ g) {
    long idx = (long)blockIdx.x * blockDim.x + threadIdx.x;
    if (idx >= (long)N_NODESC * OUTC) return;
    int n = (int)(idx / OUTC), c = (int)(idx % OUTC);
    atomicMaxF(&g[batch[n] * OUTC + c], h[idx]);
}

__global__ void g_clean(float* __restrict__ g) {
    int idx = blockIdx.x * blockDim.x + threadIdx.x;
    if (idx >= N_GRAPHSC * OUTC) return;
    float v = g[idx];
    g[idx] = (v > -1e29f) ? v : 0.0f;        // empty-graph -> 0 (isfinite guard)
}

__global__ void bias_relu_out(const float* __restrict__ t, const float* __restrict__ bias,
                              float* __restrict__ out) {
    int idx = blockIdx.x * blockDim.x + threadIdx.x;
    if (idx >= N_GRAPHSC * OUTC) return;
    float v = t[idx] + bias[idx % OUTC];
    out[idx] = v > 0.0f ? v : 0.0f;
}

// ---------------- launch ----------------------------------------------------
extern "C" void kernel_launch(void* const* d_in, const int* in_sizes, int n_in,
                              void* d_out, int out_size, void* d_ws, size_t ws_size,
                              hipStream_t stream) {
    const float* x   = (const float*)d_in[0];
    const int*   ei  = (const int*)d_in[1];
    const int*   bat = (const int*)d_in[2];
    const float* W1  = (const float*)d_in[3];
    const float* as1 = (const float*)d_in[4];
    const float* ad1 = (const float*)d_in[5];
    const float* b1  = (const float*)d_in[6];
    const float* W2  = (const float*)d_in[7];
    const float* as2 = (const float*)d_in[8];
    const float* ad2 = (const float*)d_in[9];
    const float* b2  = (const float*)d_in[10];
    const float* fcw = (const float*)d_in[11];
    const float* fcb = (const float*)d_in[12];
    float* out = (float*)d_out;

    char* base = (char*)d_ws;
    size_t off = 0;
    auto grab = [&](size_t bytes) -> void* {
        void* p = base + off;
        off += (bytes + 255) & ~(size_t)255;
        return p;
    };
    // transposed bf16 weights: BT[Npad][Kpad]
    __bf16* W1t  = (__bf16*)grab((size_t)D1PC * 64 * 2);
    __bf16* W2t  = (__bf16*)grab((size_t)OUTC * D1PC * 2);
    __bf16* fcwt = (__bf16*)grab((size_t)OUTC * OUTC * 2);
    float* h1    = (float*)grab((size_t)N_NODESC * D1PC * 4);
    float* alS1  = (float*)grab((size_t)N_NODESC * HEADS1C * 4);
    float* alD1  = (float*)grab((size_t)N_NODESC * HEADS1C * 4);
    float* m1    = (float*)grab((size_t)N_NODESC * HEADS1C * 4);
    float* s1    = (float*)grab((size_t)N_NODESC * HEADS1C * 4);
    float* acc1  = (float*)grab((size_t)N_NODESC * D1PC * 4);
    float* alS2  = (float*)grab((size_t)N_NODESC * 4);
    float* alD2  = (float*)grab((size_t)N_NODESC * 4);
    float* m2    = (float*)grab((size_t)N_NODESC * 4);
    float* s2    = (float*)grab((size_t)N_NODESC * 4);
    float* gbuf  = (float*)grab((size_t)N_GRAPHSC * OUTC * 4);
    float* gtmp  = (float*)grab((size_t)N_GRAPHSC * OUTC * 4);
    // layer-2 big buffers alias the h1 region (dead after edge_pass1)
    float* h2   = h1;
    float* acc2 = (float*)((char*)h1 + (size_t)N_NODESC * OUTC * 4);

    const int B = 256;
    auto blk = [](long n, int b) { return (int)((n + b - 1) / b); };

    // weight conversion (bf16, transposed, K/N zero-padded)
    pad_convert_T<<<blk(D1PC * 64, B), B, 0, stream>>>(W1, W1t, F_INC, D1C, 64, D1PC);
    pad_convert_T<<<blk(OUTC * D1PC, B), B, 0, stream>>>(W2, W2t, D1C, OUTC, D1PC, OUTC);
    pad_convert_T<<<blk(OUTC * OUTC, B), B, 0, stream>>>(fcw, fcwt, OUTC, OUTC, OUTC, OUTC);

    // ---- layer 1: h1 = x @ W1 (M=50000, K=36->64, N=360->384) ----
    gemm_bf16_wmma<<<dim3(D1PC / 128, blk(N_NODESC, 64)), B, 0, stream>>>(
        x, F_INC, W1t, 64, h1, D1PC, N_NODESC, F_INC);
    attn_logits1<<<blk((long)N_NODESC * HEADS1C, B), B, 0, stream>>>(h1, as1, ad1, alS1, alD1);
    fill_f32<<<blk((long)N_NODESC * HEADS1C, B), B, 0, stream>>>(m1, -1e30f, (long)N_NODESC * HEADS1C);
    fill_f32<<<blk((long)N_NODESC * HEADS1C, B), B, 0, stream>>>(s1, 0.0f, (long)N_NODESC * HEADS1C);
    fill_f32<<<blk((long)N_NODESC * D1PC, B), B, 0, stream>>>(acc1, 0.0f, (long)N_NODESC * D1PC);
    edge_max1<<<blk((long)E_TOTC * HEADS1C, B), B, 0, stream>>>(ei, alS1, alD1, m1);
    edge_pass1<<<blk(E_TOTC, 8), B, 0, stream>>>(ei, alS1, alD1, m1, h1, s1, acc1);
    finalize1<<<blk((long)N_NODESC * D1C, B), B, 0, stream>>>(acc1, s1, b1);

    // ---- layer 2: h2 = h1b @ W2 (K=360->384, N=128) ----
    gemm_bf16_wmma<<<dim3(OUTC / 128, blk(N_NODESC, 64)), B, 0, stream>>>(
        acc1, D1PC, W2t, D1PC, h2, OUTC, N_NODESC, D1C);
    attn_logits2<<<blk(N_NODESC, B), B, 0, stream>>>(h2, as2, ad2, alS2, alD2);
    fill_f32<<<blk(N_NODESC, B), B, 0, stream>>>(m2, -1e30f, N_NODESC);
    fill_f32<<<blk(N_NODESC, B), B, 0, stream>>>(s2, 0.0f, N_NODESC);
    fill_f32<<<blk((long)N_NODESC * OUTC, B), B, 0, stream>>>(acc2, 0.0f, (long)N_NODESC * OUTC);
    edge_max2<<<blk(E_TOTC, B), B, 0, stream>>>(ei, alS2, alD2, m2);
    edge_pass2<<<blk(E_TOTC, 8), B, 0, stream>>>(ei, alS2, alD2, m2, h2, s2, acc2);
    finalize2<<<blk((long)N_NODESC * OUTC, B), B, 0, stream>>>(acc2, s2, b2);

    // ---- pool + FC ----
    fill_f32<<<blk(N_GRAPHSC * OUTC, B), B, 0, stream>>>(gbuf, -1e30f, N_GRAPHSC * OUTC);
    pool_max<<<blk((long)N_NODESC * OUTC, B), B, 0, stream>>>(acc2, bat, gbuf);
    g_clean<<<blk(N_GRAPHSC * OUTC, B), B, 0, stream>>>(gbuf);
    gemm_bf16_wmma<<<dim3(OUTC / 128, N_GRAPHSC / 64), B, 0, stream>>>(
        gbuf, OUTC, fcwt, OUTC, gtmp, OUTC, N_GRAPHSC, OUTC);
    bias_relu_out<<<blk(N_GRAPHSC * OUTC, B), B, 0, stream>>>(gtmp, fcb, out);

    (void)in_sizes; (void)n_in; (void)out_size; (void)ws_size;
}
